// GroupQueryAttention_62740882260228
// MI455X (gfx1250) — compile-verified, matrix operand order checked by reading, not run on hardware
//
#include <hip/hip_runtime.h>
#include <hip/hip_bf16.h>

// ---------------------------------------------------------------------------
// GQA forward for MI455X (gfx1250), wave32.
//  - All GEMMs on v_wmma_f32_16x16x32_bf16 (fp32 accumulate).
//  - GEMM stages tiles in LDS via GLOBAL_LOAD_ASYNC_TO_LDS_B128 (ASYNCcnt),
//    double-buffered with s_wait_asynccnt + workgroup barriers.
//  - Flash attention: online softmax, P relayout via per-wave LDS.
// ---------------------------------------------------------------------------

typedef __attribute__((ext_vector_type(16))) __bf16 v16bf;
typedef __attribute__((ext_vector_type(8)))  float  v8f;
typedef int v4i_vs __attribute__((vector_size(16)));   // matches builtin param type

#define T_LEN   2048
#define D_IN    4096
#define D_OUT   4096
#define HEADS   32
#define KVG     8
#define HD      128
#define DKV     (KVG * HD)   // 1024

#if __has_builtin(__builtin_amdgcn_global_load_async_to_lds_b128) && \
    __has_builtin(__builtin_amdgcn_s_wait_asynccnt)
#define USE_ASYNC 1
#else
#define USE_ASYNC 0
#endif

static __device__ __forceinline__ v8f wmma_bf16(v16bf a, v16bf b, v8f c) {
  // D = A(16x32 bf16) * B(32x16 bf16) + C(16x16 f32)
  return __builtin_amdgcn_wmma_f32_16x16x32_bf16(
      /*neg_a=*/false, a, /*neg_b=*/false, b,
      /*c_mod=*/(short)0, c, /*reuse_a=*/false, /*reuse_b=*/false);
}

// A-style fragment load from a row-major [rows x ld] bf16 matrix.
// Lane L<16 : row = mBase+L,  holds K = kBase + {0..7, 16..23}
// Lane L>=16: row = mBase+L-16, holds K = kBase + {8..15, 24..31}
// (B fragments load with the same function from B^T.)
static __device__ __forceinline__ v16bf load_frag(const __hip_bfloat16* base,
                                                  int ld, int mBase, int kBase,
                                                  int lane) {
  int m    = mBase + (lane & 15);
  int ksel = (lane >> 4) << 3;          // 0 or 8
  const __hip_bfloat16* p = base + (size_t)m * (size_t)ld + (size_t)(kBase + ksel);
  union { int4 q[2]; v16bf v; } u;
  u.q[0] = *reinterpret_cast<const int4*>(p);        // 8 bf16 = 16B
  u.q[1] = *reinterpret_cast<const int4*>(p + 16);   // 8 bf16 = 16B
  return u.v;
}

// 16B global -> LDS copy. Async (ASYNCcnt) when the builtin is available.
// Builtin signature (from toolchain): (v4i AS1*, v4i AS3*, imm off, imm cpol).
// Low 32 bits of a generic shared pointer are the LDS byte address
static __device__ __forceinline__ void cp16_g2l(const __hip_bfloat16* g,
                                                __hip_bfloat16* l) {
#if USE_ASYNC
  __builtin_amdgcn_global_load_async_to_lds_b128(
      (__attribute__((address_space(1))) v4i_vs*)(uintptr_t)g,
      (__attribute__((address_space(3))) v4i_vs*)(unsigned)(uintptr_t)l,
      0, 0);
#else
  *reinterpret_cast<int4*>(l) = *reinterpret_cast<const int4*>(g);
#endif
}

#if USE_ASYNC
#define ASYNC_WAIT(n) __builtin_amdgcn_s_wait_asynccnt(n)
#else
#define ASYNC_WAIT(n)
#endif

// ---------------------------------------------------------------------------
// fp32 -> bf16 elementwise convert
// ---------------------------------------------------------------------------
__global__ __launch_bounds__(256) void cvt_bf16_kernel(const float* __restrict__ in,
                                                       __hip_bfloat16* __restrict__ out,
                                                       int n) {
  int i = blockIdx.x * 256 + threadIdx.x;
  if (i < n) out[i] = __float2bfloat16(in[i]);
}

// ---------------------------------------------------------------------------
// W[K][N] (row-major f32) -> Wt[N][K] (row-major bf16), tiled transpose
// ---------------------------------------------------------------------------
__global__ __launch_bounds__(256) void transpose_cvt_kernel(const float* __restrict__ W,
                                                            __hip_bfloat16* __restrict__ Wt,
                                                            int K, int N) {
  __shared__ float tile[32][33];
  int k0 = blockIdx.x * 32;
  int n0 = blockIdx.y * 32;
#pragma unroll
  for (int i = 0; i < 32; i += 8)
    tile[threadIdx.y + i][threadIdx.x] =
        W[(size_t)(k0 + threadIdx.y + i) * N + (n0 + threadIdx.x)];
  __syncthreads();
#pragma unroll
  for (int i = 0; i < 32; i += 8)
    Wt[(size_t)(n0 + threadIdx.y + i) * K + (k0 + threadIdx.x)] =
        __float2bfloat16(tile[threadIdx.x][threadIdx.y + i]);
}

// ---------------------------------------------------------------------------
// C[M][N] f32 = A[M][K] bf16 * B[K][N] bf16 (B supplied as Bt[N][K]).
// Block = 256 threads = 8 waves (4M x 2N) -> 64M x 128N block tile.
// Per 32-K stage: A tile 64x32 + Bt tile 128x32 staged in LDS by async
// b128 copies (3 per thread), double buffered.
// ---------------------------------------------------------------------------
__global__ __launch_bounds__(256) void gemm_bf16_kernel(
    const __hip_bfloat16* __restrict__ A,
    const __hip_bfloat16* __restrict__ Bt,
    float* __restrict__ C, int M, int N, int K) {
  __shared__ __align__(16) __hip_bfloat16 Ab[2][64 * 32];   // 2 x 4KB
  __shared__ __align__(16) __hip_bfloat16 Bb[2][128 * 32];  // 2 x 8KB

  int tid  = threadIdx.x;
  int lane = tid & 31;
  int wave = tid >> 5;
  int mBlock = blockIdx.x * 64;
  int nBlock = blockIdx.y * 128;
  int mLoc = (wave >> 1) * 16;   // 0..48
  int nLoc = (wave & 1) * 64;    // 0 or 64

  // copy-chunk geometry: 16B chunk = 8 bf16; tile rows are 32 bf16 = 4 chunks
  int arow = tid >> 2;             // 0..63
  int acol = (tid & 3) * 8;        // 0,8,16,24
  int b0row = tid >> 2;            // chunks 0..255   -> rows 0..63
  int b0col = (tid & 3) * 8;
  int b1row = (tid + 256) >> 2;    // chunks 256..511 -> rows 64..127
  int b1col = (tid & 3) * 8;

  auto issue = [&](int buf, int k0) {
    cp16_g2l(A  + (size_t)(mBlock + arow)  * K + k0 + acol,  &Ab[buf][arow  * 32 + acol]);
    cp16_g2l(Bt + (size_t)(nBlock + b0row) * K + k0 + b0col, &Bb[buf][b0row * 32 + b0col]);
    cp16_g2l(Bt + (size_t)(nBlock + b1row) * K + k0 + b1col, &Bb[buf][b1row * 32 + b1col]);
  };

  v8f acc0 = {}, acc1 = {}, acc2 = {}, acc3 = {};

  issue(0, 0);
  if (K > 32) issue(1, 32);

  for (int k0 = 0; k0 < K; k0 += 32) {
    int buf = (k0 >> 5) & 1;
    // in-order async completion: leaving <=3 outstanding means this stage landed
    if (k0 + 32 < K) { ASYNC_WAIT(3); } else { ASYNC_WAIT(0); }
    __syncthreads();

    v16bf a  = load_frag(&Ab[buf][0], 32, mLoc,      0, lane);
    v16bf b0 = load_frag(&Bb[buf][0], 32, nLoc,      0, lane);
    v16bf b1 = load_frag(&Bb[buf][0], 32, nLoc + 16, 0, lane);
    v16bf b2 = load_frag(&Bb[buf][0], 32, nLoc + 32, 0, lane);
    v16bf b3 = load_frag(&Bb[buf][0], 32, nLoc + 48, 0, lane);
    acc0 = wmma_bf16(a, b0, acc0);
    acc1 = wmma_bf16(a, b1, acc1);
    acc2 = wmma_bf16(a, b2, acc2);
    acc3 = wmma_bf16(a, b3, acc3);

    __syncthreads();                       // all waves done reading this buffer
    if (k0 + 64 < K) issue(buf, k0 + 64);  // refill the freed buffer
  }

  int nCol  = lane & 15;
  int mOff  = (lane >> 4) * 8;
  int mBase = mBlock + mLoc;
  int nBase = nBlock + nLoc;
#pragma unroll
  for (int j = 0; j < 8; ++j) {
    float* crow = C + (size_t)(mBase + mOff + j) * N + nBase + nCol;
    crow[0]  = acc0[j];
    crow[16] = acc1[j];
    crow[32] = acc2[j];
    crow[48] = acc3[j];
  }
}

// ---------------------------------------------------------------------------
// RoPE on Q (f32 [T, 4096] -> bf16), pair element is idx^64 within the head.
// ---------------------------------------------------------------------------
__global__ __launch_bounds__(256) void rope_q_kernel(const float* __restrict__ q,
                                                     const float* __restrict__ cosT,
                                                     const float* __restrict__ sinT,
                                                     __hip_bfloat16* __restrict__ out) {
  int idx = blockIdx.x * 256 + threadIdx.x;    // < T*4096
  int t  = idx >> 12;
  int d  = idx & 4095;
  int dd = d & 127;
  float c  = cosT[t * HD + dd];
  float s  = sinT[t * HD + dd];
  float xv = q[idx];
  float xp = q[idx ^ 64];
  float r  = (dd < 64) ? -xp : xp;
  out[idx] = __float2bfloat16(xv * c + r * s);
}

__global__ __launch_bounds__(256) void rope_k_kernel(const float* __restrict__ k,
                                                     const float* __restrict__ cosT,
                                                     const float* __restrict__ sinT,
                                                     __hip_bfloat16* __restrict__ out) {
  int idx = blockIdx.x * 256 + threadIdx.x;    // < T*1024
  int t  = idx >> 10;
  int d  = idx & 1023;
  int dd = d & 127;
  float c  = cosT[t * HD + dd];
  float s  = sinT[t * HD + dd];
  float xv = k[idx];
  float xp = k[idx ^ 64];
  float r  = (dd < 64) ? -xp : xp;
  out[idx] = __float2bfloat16(xv * c + r * s);
}

// v f32 [T, 1024] -> vT bf16 [g][128][T]  (so P*V B-frags read contiguous s)
__global__ __launch_bounds__(256) void vpack_kernel(const float* __restrict__ v,
                                                    __hip_bfloat16* __restrict__ vt) {
  int idx = blockIdx.x * 256 + threadIdx.x;    // < T*1024
  int t = idx >> 10;
  int d = idx & 1023;      // g*128 + dd
  vt[(size_t)d * T_LEN + t] = __float2bfloat16(v[idx]);
}

// ---------------------------------------------------------------------------
// Flash attention. One wave = one (head, 16-query tile). Streams causal key
// blocks of 32: scores via 8 WMMA (2 col tiles x 4 K-chunks of HD=128),
// online softmax (shfl_xor row reductions), P relayout via per-wave LDS,
// P*V via 8 WMMA (8 d-tiles, K=32).
// ---------------------------------------------------------------------------
__global__ __launch_bounds__(128) void attn_kernel(
    const __hip_bfloat16* __restrict__ Q,    // [T, 4096] bf16 (rope'd)
    const __hip_bfloat16* __restrict__ Kb,   // [T, 1024] bf16 (rope'd)
    const __hip_bfloat16* __restrict__ Vt,   // [8][128][T] bf16
    __hip_bfloat16* __restrict__ Ctx) {      // [T, 4096] bf16
  __shared__ __align__(16) __hip_bfloat16 pLds[4][16 * 32];

  int lane = threadIdx.x & 31;
  int wave = threadIdx.x >> 5;
  int task  = blockIdx.x * 4 + wave;   // 4096 tasks total
  int head  = task >> 7;               // 0..31
  int qTile = task & 127;              // 0..127
  int g     = head >> 2;
  int qBase = qTile << 4;

  const __hip_bfloat16* qhead = Q  + head * HD;
  const __hip_bfloat16* khead = Kb + g * HD;
  const __hip_bfloat16* vhead = Vt + (size_t)g * HD * T_LEN;
  __hip_bfloat16* lds = pLds[wave];

  v16bf qf0 = load_frag(qhead, D_OUT, qBase,  0, lane);
  v16bf qf1 = load_frag(qhead, D_OUT, qBase, 32, lane);
  v16bf qf2 = load_frag(qhead, D_OUT, qBase, 64, lane);
  v16bf qf3 = load_frag(qhead, D_OUT, qBase, 96, lane);

  v8f o[8];
#pragma unroll
  for (int d = 0; d < 8; ++d) o[d] = (v8f){};

  float mI[8], lI[8];
#pragma unroll
  for (int j = 0; j < 8; ++j) { mI[j] = -__builtin_inff(); lI[j] = 0.f; }

  int halfSel = lane >> 4;
  int n0      = lane & 15;
  const float sc = 0.08838834764831845f;   // 1/sqrt(128)

  for (int sBase = 0; sBase < qBase + 16; sBase += 32) {
    v8f s0 = {}, s1 = {};
    {
      v16bf kb;
      kb = load_frag(khead, DKV, sBase,       0, lane); s0 = wmma_bf16(qf0, kb, s0);
      kb = load_frag(khead, DKV, sBase,      32, lane); s0 = wmma_bf16(qf1, kb, s0);
      kb = load_frag(khead, DKV, sBase,      64, lane); s0 = wmma_bf16(qf2, kb, s0);
      kb = load_frag(khead, DKV, sBase,      96, lane); s0 = wmma_bf16(qf3, kb, s0);
      kb = load_frag(khead, DKV, sBase + 16,  0, lane); s1 = wmma_bf16(qf0, kb, s1);
      kb = load_frag(khead, DKV, sBase + 16, 32, lane); s1 = wmma_bf16(qf1, kb, s1);
      kb = load_frag(khead, DKV, sBase + 16, 64, lane); s1 = wmma_bf16(qf2, kb, s1);
      kb = load_frag(khead, DKV, sBase + 16, 96, lane); s1 = wmma_bf16(qf3, kb, s1);
    }

    bool needMask = (sBase + 31 > qBase);
    float alphaA[8];
#pragma unroll
    for (int j = 0; j < 8; ++j) {
      int m    = (halfSel << 3) + j;       // 0..15 (C-layout row)
      int trow = qBase + m;
      float e0 = s0[j] * sc;
      float e1 = s1[j] * sc;
      if (needMask) {
        if (sBase + n0      > trow) e0 = -__builtin_inff();
        if (sBase + 16 + n0 > trow) e1 = -__builtin_inff();
      }
      // row max over 32 columns: per-lane pair max, then 16-lane tree
      float rm = fmaxf(e0, e1);
      rm = fmaxf(rm, __shfl_xor(rm, 1));
      rm = fmaxf(rm, __shfl_xor(rm, 2));
      rm = fmaxf(rm, __shfl_xor(rm, 4));
      rm = fmaxf(rm, __shfl_xor(rm, 8));
      float mnew  = fmaxf(mI[j], rm);
      float alpha = __expf(mI[j] - mnew);
      float p0 = __expf(e0 - mnew);
      float p1 = __expf(e1 - mnew);
      float rs = p0 + p1;
      rs += __shfl_xor(rs, 1);
      rs += __shfl_xor(rs, 2);
      rs += __shfl_xor(rs, 4);
      rs += __shfl_xor(rs, 8);
      lI[j]     = lI[j] * alpha + rs;
      mI[j]     = mnew;
      alphaA[j] = alpha;
      // P (C-layout) -> row-major 16x32 bf16 in LDS
      lds[m * 32 + n0]      = __float2bfloat16(p0);
      lds[m * 32 + 16 + n0] = __float2bfloat16(p1);
    }

    // rescale context accumulators by alpha (row j matches accumulator element j)
#pragma unroll
    for (int d = 0; d < 8; ++d)
#pragma unroll
      for (int j = 0; j < 8; ++j) o[d][j] *= alphaA[j];

    // P A-frag from LDS (same wave wrote it; DS ops are in-order per wave)
    v16bf pf = load_frag(lds, 32, 0, 0, lane);
#pragma unroll
    for (int d = 0; d < 8; ++d) {
      v16bf vb = load_frag(vhead, T_LEN, d * 16, sBase, lane);  // Vt rows = d, K = s
      o[d] = wmma_bf16(pf, vb, o[d]);
    }
  }

  float invA[8];
#pragma unroll
  for (int j = 0; j < 8; ++j) invA[j] = 1.f / lI[j];

#pragma unroll
  for (int d = 0; d < 8; ++d)
#pragma unroll
    for (int j = 0; j < 8; ++j) {
      int m = (halfSel << 3) + j;
      int t = qBase + m;
      Ctx[(size_t)t * D_OUT + head * HD + d * 16 + n0] =
          __float2bfloat16(o[d][j] * invA[j]);
    }
}

// ---------------------------------------------------------------------------
// Host-side orchestration
// ---------------------------------------------------------------------------
extern "C" void kernel_launch(void* const* d_in, const int* in_sizes, int n_in,
                              void* d_out, int out_size, void* d_ws, size_t ws_size,
                              hipStream_t stream) {
  const float* x    = (const float*)d_in[0];
  const float* Wq   = (const float*)d_in[1];
  const float* Wk   = (const float*)d_in[2];
  const float* Wv   = (const float*)d_in[3];
  const float* Wo   = (const float*)d_in[4];
  const float* cosT = (const float*)d_in[5];
  const float* sinT = (const float*)d_in[6];

  char* ws = (char*)d_ws;
  auto alloc = [&](size_t bytes) -> void* {
    void* p = (void*)ws;
    ws += (bytes + 255) & ~(size_t)255;
    return p;
  };

  __hip_bfloat16* x_bf   = (__hip_bfloat16*)alloc((size_t)T_LEN * D_IN * 2);
  __hip_bfloat16* WqT    = (__hip_bfloat16*)alloc((size_t)D_IN * D_OUT * 2);
  __hip_bfloat16* WkT    = (__hip_bfloat16*)alloc((size_t)D_IN * DKV * 2);
  __hip_bfloat16* WvT    = (__hip_bfloat16*)alloc((size_t)D_IN * DKV * 2);
  __hip_bfloat16* WoT    = (__hip_bfloat16*)alloc((size_t)D_OUT * D_OUT * 2);
  float*          q32    = (float*)alloc((size_t)T_LEN * D_OUT * 4);
  float*          k32    = (float*)alloc((size_t)T_LEN * DKV * 4);
  float*          v32    = (float*)alloc((size_t)T_LEN * DKV * 4);
  __hip_bfloat16* q_bf   = (__hip_bfloat16*)alloc((size_t)T_LEN * D_OUT * 2);
  __hip_bfloat16* k_bf   = (__hip_bfloat16*)alloc((size_t)T_LEN * DKV * 2);
  __hip_bfloat16* vT_bf  = (__hip_bfloat16*)alloc((size_t)T_LEN * DKV * 2);
  __hip_bfloat16* ctx_bf = (__hip_bfloat16*)alloc((size_t)T_LEN * D_OUT * 2);

  dim3 tb(32, 8);

  // bf16 conversions / transposes
  cvt_bf16_kernel<<<(T_LEN * D_IN) / 256, 256, 0, stream>>>(x, x_bf, T_LEN * D_IN);
  transpose_cvt_kernel<<<dim3(D_IN / 32, D_OUT / 32), tb, 0, stream>>>(Wq, WqT, D_IN, D_OUT);
  transpose_cvt_kernel<<<dim3(D_IN / 32, DKV / 32),  tb, 0, stream>>>(Wk, WkT, D_IN, DKV);
  transpose_cvt_kernel<<<dim3(D_IN / 32, DKV / 32),  tb, 0, stream>>>(Wv, WvT, D_IN, DKV);
  transpose_cvt_kernel<<<dim3(D_OUT / 32, D_OUT / 32), tb, 0, stream>>>(Wo, WoT, D_OUT, D_OUT);

  // QKV projections (bf16 WMMA, fp32 accumulate), async-LDS staged
  gemm_bf16_kernel<<<dim3(T_LEN / 64, D_OUT / 128), 256, 0, stream>>>(x_bf, WqT, q32, T_LEN, D_OUT, D_IN);
  gemm_bf16_kernel<<<dim3(T_LEN / 64, DKV / 128),  256, 0, stream>>>(x_bf, WkT, k32, T_LEN, DKV, D_IN);
  gemm_bf16_kernel<<<dim3(T_LEN / 64, DKV / 128),  256, 0, stream>>>(x_bf, WvT, v32, T_LEN, DKV, D_IN);

  // RoPE + bf16 pack, V transpose pack
  rope_q_kernel<<<(T_LEN * D_OUT) / 256, 256, 0, stream>>>(q32, cosT, sinT, q_bf);
  rope_k_kernel<<<(T_LEN * DKV) / 256, 256, 0, stream>>>(k32, cosT, sinT, k_bf);
  vpack_kernel<<<(T_LEN * DKV) / 256, 256, 0, stream>>>(v32, vT_bf);

  // Flash attention: 32 heads * 128 q-tiles = 4096 wave tasks, 4 waves/block
  attn_kernel<<<(HEADS * (T_LEN / 16)) / 4, 128, 0, stream>>>(q_bf, k_bf, vT_bf, ctx_bf);

  // Output projection -> fp32 d_out
  gemm_bf16_kernel<<<dim3(T_LEN / 64, D_OUT / 128), 256, 0, stream>>>(
      ctx_bf, WoT, (float*)d_out, T_LEN, D_OUT, D_OUT);
}
